// CMIAttentionMatrixForAcrobot_38019050504386
// MI455X (gfx1250) — compile-verified, matrix operand order checked by reading, not run on hardware
//
#include <hip/hip_runtime.h>
#include <hip/hip_bf16.h>
#include <math.h>

typedef __attribute__((ext_vector_type(16))) __bf16 v16bf;
typedef __attribute__((ext_vector_type(8)))  float  v8f;

#define BM 128
#define BN 128
#define BK 32
#define LDS_STRIDE 40   // 32 bf16 + 8 pad -> 80B row stride (conflict-free frag reads)
#define NTHREADS 256

// f32 pair -> packed bf16x2 (RNE) using v_perm_b32 to splice the high halves.
__device__ __forceinline__ unsigned pack_bf16x2(float a, float b) {
  unsigned ua = __float_as_uint(a), ub = __float_as_uint(b);
  ua += 0x7FFFu + ((ua >> 16) & 1u);
  ub += 0x7FFFu + ((ub >> 16) & 1u);
  return __builtin_amdgcn_perm(ub, ua, 0x07060302);  // {ub.hi16, ua.hi16}
}

__device__ __forceinline__ unsigned short f32_to_bf16(float a) {
  unsigned ua = __float_as_uint(a);
  ua += 0x7FFFu + ((ua >> 16) & 1u);
  return (unsigned short)(ua >> 16);
}

__device__ __forceinline__ float bf16_to_f32(unsigned short h) {
  return __uint_as_float(((unsigned)h) << 16);
}

// ---- CDNA5 async global->LDS (ASYNCcnt tracked), inline asm (toolchain-portable) ----
__device__ __forceinline__ void async_load_b128(const unsigned short* gsrc, unsigned lds_off) {
  unsigned long long ga = (unsigned long long)(uintptr_t)gsrc;
  asm volatile("global_load_async_to_lds_b128 %0, %1, off"
               :: "v"(lds_off), "v"(ga) : "memory");
}

template <int N>
__device__ __forceinline__ void wait_asynccnt() {
  asm volatile("s_wait_asynccnt %0" :: "n"(N) : "memory");
}

// Issue one [128 x 32] bf16 tile: 512 16B chunks, 2 per thread -> 4 async instr
// per wave per (A,B) tile pair when called twice.
__device__ __forceinline__ void issue_tile(unsigned short* sh, const unsigned short* __restrict__ g,
                                           int row0, int k0, int ld) {
  int t = threadIdx.x;
#pragma unroll
  for (int h = 0; h < 2; ++h) {
    int chunk = t + h * 256;       // 0..511
    int row   = chunk >> 2;        // 0..127
    int kc    = (chunk & 3) * 8;   // 0,8,16,24
    const unsigned short* gp = g + (size_t)(row0 + row) * ld + k0 + kc;
    unsigned lo = (unsigned)(uintptr_t)(sh + row * LDS_STRIDE + kc);
    async_load_b128(gp, lo);
  }
}

// WMMA 16x32 bf16 fragment from LDS (rows = M for A / N for B, K contiguous).
// lanes 0-15: row=lane,    halves 0..7 = K0..7,  8..15 = K16..23
// lanes 16-31: row=lane-16, halves 0..7 = K8..15, 8..15 = K24..31
__device__ __forceinline__ v16bf load_frag(const unsigned short* sh, int rowBase) {
  int lane = threadIdx.x & 31;
  int r = rowBase + (lane & 15);
  int k = (lane >> 4) * 8;
  union { v16bf v; uint4 q[2]; } u;
  const unsigned short* p = sh + r * LDS_STRIDE + k;
  u.q[0] = *(const uint4*)(p);
  u.q[1] = *(const uint4*)(p + 16);
  return u.v;
}

// NT GEMM on pre-converted bf16: C = A[.,K] * B[.,K]^T * scale (+ biasN[col]).
// A/B row-padded to tile multiples by construction (no bounds logic in loop).
// Stores clipped to gm < Cm, gn < Cn; optional f32 and/or bf16 outputs.
__global__ __launch_bounds__(NTHREADS)
void wmma_nt_gemm(const unsigned short* __restrict__ A, const unsigned short* __restrict__ B,
                  float* __restrict__ Cf, unsigned short* __restrict__ Cb,
                  int K, int ldc, int Cm, int Cn,
                  const float* __restrict__ biasN, float scale) {
  __shared__ unsigned short shA[2][BM * LDS_STRIDE];
  __shared__ unsigned short shB[2][BN * LDS_STRIDE];

  int tileM = blockIdx.y * BM;
  int tileN = blockIdx.x * BN;

  int wave  = threadIdx.x >> 5;
  int waveM = wave & 3;   // 4 x 32 rows = 128
  int waveN = wave >> 2;  // 2 x 64 cols = 128

  v8f acc[2][4];
#pragma unroll
  for (int i = 0; i < 2; ++i)
#pragma unroll
    for (int j = 0; j < 4; ++j)
      acc[i][j] = (v8f){0.f, 0.f, 0.f, 0.f, 0.f, 0.f, 0.f, 0.f};

  const int T = K / BK;
  // Preload tile 0 into buffer 0.
  issue_tile(shA[0], A, tileM, 0, K);
  issue_tile(shB[0], B, tileN, 0, K);

  for (int kt = 0; kt < T; ++kt) {
    int c = kt & 1;
    if (kt + 1 < T) {
      // Lookahead: buffer c^1 was last read at iter kt-1; end-of-iter barrier
      // guarantees those reads are done before these writes can land.
      issue_tile(shA[c ^ 1], A, tileM, (kt + 1) * BK, K);
      issue_tile(shB[c ^ 1], B, tileN, (kt + 1) * BK, K);
      wait_asynccnt<4>();   // 4 instr/wave/tile-pair in flight for kt+1 -> tile kt landed
    } else {
      wait_asynccnt<0>();   // drain
    }
    __syncthreads();        // all waves' portions of tile kt are in LDS

    v16bf af[2], bfr[4];
#pragma unroll
    for (int mi = 0; mi < 2; ++mi) af[mi] = load_frag(shA[c], waveM * 32 + mi * 16);
#pragma unroll
    for (int ni = 0; ni < 4; ++ni) bfr[ni] = load_frag(shB[c], waveN * 64 + ni * 16);

#pragma unroll
    for (int mi = 0; mi < 2; ++mi)
#pragma unroll
      for (int ni = 0; ni < 4; ++ni)
        acc[mi][ni] = __builtin_amdgcn_wmma_f32_16x16x32_bf16(
            false, af[mi], false, bfr[ni], (short)0, acc[mi][ni], false, false);

    __syncthreads();        // everyone done reading buffer c before it is overwritten
  }

  // Epilogue. D layout: VGPR r, lane L: M = r + (L>=16 ? 8 : 0), N = L & 15.
  int lane = threadIdx.x & 31;
  int cn   = lane & 15;
  int rOff = (lane >> 4) * 8;
#pragma unroll
  for (int mi = 0; mi < 2; ++mi)
#pragma unroll
    for (int ni = 0; ni < 4; ++ni)
#pragma unroll
      for (int r = 0; r < 8; ++r) {
        int gm = tileM + waveM * 32 + mi * 16 + rOff + r;
        int gn = tileN + waveN * 64 + ni * 16 + cn;
        if (gm < Cm && gn < Cn) {
          float v = acc[mi][ni][r] * scale;
          if (biasN) v += biasN[gn];
          if (Cf) Cf[(size_t)gm * ldc + gn] = v;
          if (Cb) Cb[(size_t)gm * ldc + gn] = f32_to_bf16(v);
        }
      }
}

// Bulk f32 -> bf16 (n multiple of 8).
__global__ __launch_bounds__(256)
void cvt_kernel(const float* __restrict__ src, unsigned short* __restrict__ dst, int n) {
  int i = (blockIdx.x * blockDim.x + threadIdx.x) * 8;
  if (i >= n) return;
  float4 a = *(const float4*)(src + i);
  float4 b = *(const float4*)(src + i + 4);
  uint4 o;
  o.x = pack_bf16x2(a.x, a.y);
  o.y = pack_bf16x2(a.z, a.w);
  o.z = pack_bf16x2(b.x, b.y);
  o.w = pack_bf16x2(b.z, b.w);
  *(uint4*)(dst + i) = o;
}

// f32 [rows_valid,4096] -> bf16 [rows_pad,4096], zero-filled pad rows.
__global__ __launch_bounds__(256)
void cvt_pad_kernel(const float* __restrict__ src, unsigned short* __restrict__ dst,
                    int rows_valid) {
  int i = (blockIdx.x * blockDim.x + threadIdx.x) * 2;
  int row = i >> 12;
  float a = 0.f, b = 0.f;
  if (row < rows_valid) { a = src[i]; b = src[i + 1]; }
  *(unsigned*)(dst + i) = pack_bf16x2(a, b);
}

// context[j,n] = sum_m W_lin[j,m] * q_bf16[m,n] + b_lin[j]   ([6,4096])
// q reads coalesced across n; W_lin/b_lin addresses wave-uniform -> SMEM loads.
__global__ __launch_bounds__(256)
void context_kernel(const unsigned short* __restrict__ qb, const float* __restrict__ Wl,
                    const float* __restrict__ bl, float* __restrict__ ctx) {
  int n = blockIdx.x * blockDim.x + threadIdx.x;  // 0..4095
  float acc[6] = {0.f, 0.f, 0.f, 0.f, 0.f, 0.f};
  for (int m = 0; m < 4096; ++m) {
    float qv = bf16_to_f32(qb[(size_t)m * 4096 + n]);
#pragma unroll
    for (int j = 0; j < 6; ++j) acc[j] += Wl[j * 4096 + m] * qv;
  }
#pragma unroll
  for (int j = 0; j < 6; ++j) ctx[j * 4096 + n] = acc[j] + bl[j];
}

// k_mod = relu6(k^2 + 2k + ctx*(1+|k|)) -> bf16, padded to [128,4096] with zeros.
__global__ __launch_bounds__(256)
void kmod_kernel(const float* __restrict__ k, const float* __restrict__ ctx,
                 unsigned short* __restrict__ kmodb) {
  int i = blockIdx.x * blockDim.x + threadIdx.x;  // 128*4096
  int j = i >> 12, m = i & 4095;
  float v = 0.f;
  if (j < 6) {
    float kv = k[j * 4096 + m];
    float c  = ctx[j * 4096 + m];
    float x  = kv * kv + 2.f * kv + c * (1.f + fabsf(kv));
    v = fminf(fmaxf(x, 0.f), 6.f);
  }
  kmodb[i] = f32_to_bf16(v);
}

extern "C" void kernel_launch(void* const* d_in, const int* in_sizes, int n_in,
                              void* d_out, int out_size, void* d_ws, size_t ws_size,
                              hipStream_t stream) {
  const float* data_q = (const float*)d_in[0];  // [4096,4096]
  const float* data_k = (const float*)d_in[1];  // [6,4096]
  const float* W_q    = (const float*)d_in[2];  // [4096,4096]
  const float* b_q    = (const float*)d_in[3];  // [4096]
  const float* W_lin  = (const float*)d_in[4];  // [6,4096]
  const float* b_lin  = (const float*)d_in[5];  // [6]
  const float* W_k    = (const float*)d_in[6];  // [4096,4096]
  const float* b_k    = (const float*)d_in[7];  // [4096]
  float* out = (float*)d_out;                   // [4096,6]

  const int D = 4096;
  const size_t DD = (size_t)D * D;

  unsigned short* dqb   = (unsigned short*)d_ws;       // bf16(data_q)      32 MB
  unsigned short* wqb   = dqb + DD;                    // bf16(W_q)         32 MB
  unsigned short* qb    = wqb + DD;                    // bf16(q)           32 MB
  unsigned short* wkb   = qb + DD;                     // bf16(W_k)         32 MB
  unsigned short* dkb   = wkb + DD;                    // bf16(data_k) pad  [128,4096]
  unsigned short* kmodb = dkb + 128 * D;               // bf16(k_mod)  pad  [128,4096]
  float* k_f32   = (float*)(kmodb + 128 * D);          // k            [6,4096]
  float* ctx_f32 = k_f32 + 6 * D;                      // context      [6,4096]

  dim3 blk(NTHREADS);

  // 0) one-shot bf16 conversions (bandwidth-bound; results live in 192MB L2)
  cvt_kernel<<<DD / 8 / 256, blk, 0, stream>>>(data_q, dqb, (int)DD);
  cvt_kernel<<<DD / 8 / 256, blk, 0, stream>>>(W_q,    wqb, (int)DD);
  cvt_kernel<<<DD / 8 / 256, blk, 0, stream>>>(W_k,    wkb, (int)DD);
  cvt_pad_kernel<<<128 * D / 2 / 256, blk, 0, stream>>>(data_k, dkb, 6);

  // 1) q = data_q @ W_q^T + b_q  -> bf16 [4096,4096]
  {
    dim3 grid(D / BN, D / BM);
    wmma_nt_gemm<<<grid, blk, 0, stream>>>(dqb, wqb, nullptr, qb,
                                           D, D, D, D, b_q, 1.0f);
  }
  // 2) context = W_lin @ q + b_lin   [6,4096]
  context_kernel<<<D / 256, blk, 0, stream>>>(qb, W_lin, b_lin, ctx_f32);
  // 3) k = data_k @ W_k^T + b_k      [6,4096] f32
  {
    dim3 grid(D / BN, 1);
    wmma_nt_gemm<<<grid, blk, 0, stream>>>(dkb, wkb, k_f32, nullptr,
                                           D, D, 6, D, b_k, 1.0f);
  }
  // 4) k_mod -> bf16 padded [128,4096]
  kmod_kernel<<<128 * D / 256, blk, 0, stream>>>(k_f32, ctx_f32, kmodb);
  // 5) dot = q @ k_mod^T / sqrt(4096)  [4096,6]
  {
    dim3 grid(1, D / BM);
    wmma_nt_gemm<<<grid, blk, 0, stream>>>(qb, kmodb, out, nullptr,
                                           D, 6, D, 6, nullptr, 1.0f / 64.0f);
  }
}
